// Attention_90245852823971
// MI455X (gfx1250) — matrix operand order repaired and checked by reading, not compiled, not run on hardware
//
#include <hip/hip_runtime.h>

// ---------------------------------------------------------------------------
// Multi-head attention forward for MI455X (gfx1250, wave32, WMMA).
//   - all matmuls via v_wmma_f32_16x16x32_bf16 (fp32 accumulate)
//   - GEMMs: per-block 16x1024 A panel staged to LDS via
//     global_load_async_to_lds_b128 (ASYNCcnt), 16x64 register blocking/wave
//   - attention: 16x2048 fp32 scores + bf16 probs in 192KB LDS per WGP
// ---------------------------------------------------------------------------

#define B_  4
#define N_  2048
#define C_  1024
#define H_  16
#define D_  64
#define SCALE_ 0.125f   // HEAD_DIM^-0.5

typedef __attribute__((ext_vector_type(16))) __bf16 v16bf;
typedef __attribute__((ext_vector_type(8)))  float  v8f;
typedef int v4i_ __attribute__((vector_size(16)));
typedef __attribute__((address_space(1))) v4i_ GV4;
typedef __attribute__((address_space(3))) v4i_ LV4;

union Frag16 {
  v16bf v;
  unsigned short u[16];
  uint4 q[2];
};

#if defined(__has_builtin)
#  if __has_builtin(__builtin_amdgcn_global_load_async_to_lds_b128) && \
      __has_builtin(__builtin_amdgcn_s_wait_asynccnt)
#    define HAVE_ASYNC_LDS 1
#  endif
#endif
#ifndef HAVE_ASYNC_LDS
#  define HAVE_ASYNC_LDS 0
#endif

static __device__ __forceinline__ unsigned short f2bf(float f) {
  unsigned int u = __float_as_uint(f);
  u += 0x7FFFu + ((u >> 16) & 1u);      // round-to-nearest-even
  return (unsigned short)(u >> 16);
}

// Element e (0..15) of a 16-bit A/B WMMA fragment -> K offset in 32-wide chunk
// (ISA 7.12.2): per lane, e=0..7 and e=8..15 are contiguous 8-element runs.
static __device__ __forceinline__ int kmap(int e, int hv) {
  return (e & 7) + ((e >> 3) << 4) + (hv << 3);
}

// A-fragment from a row-major bf16 row pointer: two 16-byte loads.
static __device__ __forceinline__ void loadA(Frag16& f,
                                             const unsigned short* __restrict__ row,
                                             int k0, int hv) {
  const unsigned short* p = row + k0 + (hv << 3);
  f.q[0] = *(const uint4*)(p);
  f.q[1] = *(const uint4*)(p + 16);
}

// B-fragment from fragment-major storage [tile][lane][16]: two 16-byte loads.
static __device__ __forceinline__ void loadBf(Frag16& f,
                                              const unsigned short* __restrict__ base,
                                              size_t tile, int lane) {
  const unsigned short* p = base + (tile * 32 + lane) * 16;
  f.q[0] = ((const uint4*)p)[0];
  f.q[1] = ((const uint4*)p)[1];
}

// Cooperative panel stage (global -> LDS) in 16-byte chunks.
static __device__ __forceinline__ void stage_panel(
    const unsigned short* __restrict__ g, unsigned short* __restrict__ l,
    int chunks, int tid, int nthreads) {
#if HAVE_ASYNC_LDS
  for (int i = tid; i < chunks; i += nthreads)
    __builtin_amdgcn_global_load_async_to_lds_b128(
        (GV4*)(g + i * 8), (LV4*)(l + i * 8), 0, 0);
  __builtin_amdgcn_s_wait_asynccnt(0);
#else
  for (int i = tid; i < chunks; i += nthreads)
    *(uint4*)(l + i * 8) = *(const uint4*)(g + i * 8);
#endif
}

// ---------------------------------------------------------------------------
__global__ void cvt_bf16(const float* __restrict__ src,
                         unsigned short* __restrict__ dst, int n) {
  for (int i = blockIdx.x * blockDim.x + threadIdx.x; i < n;
       i += gridDim.x * blockDim.x)
    dst[i] = f2bf(src[i]);
}

// fp32 row-major [K][Ncols] -> bf16 fragment-major [kt][nt][lane][16]
__global__ void repack_w_frag(const float* __restrict__ w,
                              unsigned short* __restrict__ out,
                              int K, int Ncols) {
  const int NT = Ncols >> 4;
  const int total = K * Ncols;
  for (int idx = blockIdx.x * blockDim.x + threadIdx.x; idx < total;
       idx += gridDim.x * blockDim.x) {
    const int e    = idx & 15;
    const int lane = (idx >> 4) & 31;
    const int tile = idx >> 9;
    const int nt   = tile % NT;
    const int kt   = tile / NT;
    const int hv   = lane >> 4, ln = lane & 15;
    out[idx] = f2bf(w[(size_t)(kt * 32 + kmap(e, hv)) * Ncols + nt * 16 + ln]);
  }
}

// V (B,H,N,D) bf16 -> fragment-major [bh][kt(64)][nt(4)][lane][16]
__global__ void repack_v(const unsigned short* __restrict__ Vh,
                         unsigned short* __restrict__ Vf) {
  const int total = B_ * H_ * N_ * D_;
  for (int idx = blockIdx.x * blockDim.x + threadIdx.x; idx < total;
       idx += gridDim.x * blockDim.x) {
    const int e    = idx & 15;
    const int lane = (idx >> 4) & 31;
    const int nt   = (idx >> 9) & 3;
    const int kt   = (idx >> 11) & 63;
    const int bh   = idx >> 17;
    const int hv   = lane >> 4, ln = lane & 15;
    Vf[idx] = Vh[((size_t)bh * N_ + kt * 32 + kmap(e, hv)) * D_ + nt * 16 + ln];
  }
}

// ---------------------------------------------------------------------------
// QKV projection: [8192 x 1024] x [1024 x 3072] -> Q,K,V (B,H,N,D) bf16.
// Block = 16 rows x 512 cols; A panel (32 KB) in LDS; wave = 16x64 strip.
__global__ __launch_bounds__(256) void qkv_gemm(
    const unsigned short* __restrict__ xb,
    const unsigned short* __restrict__ wf,
    unsigned short* __restrict__ Qh,
    unsigned short* __restrict__ Kh,
    unsigned short* __restrict__ Vh) {
  __shared__ unsigned short Apan[16 * C_];         // 32 KB
  const int tid = threadIdx.x;
  const int grp = blockIdx.x % 6;                  // 3072 / 512 col groups
  const int tm  = blockIdx.x / 6;
  const int m0  = tm * 16;

  stage_panel(xb + (size_t)m0 * C_, Apan, (16 * C_) / 8, tid, 256);
  __syncthreads();

  const int wave = tid >> 5, lane = tid & 31;
  const int hv = lane >> 4, ln = lane & 15;
  const int n0  = grp * 512 + wave * 64;
  const int NT  = (3 * C_) / 16;                   // 192
  const int tn0 = n0 >> 4;

  v8f acc[4] = {};
  for (int kt = 0; kt < C_ / 32; ++kt) {
    Frag16 a;
    const unsigned short* ap = Apan + ln * C_ + kt * 32 + (hv << 3);
    a.q[0] = ((const uint4*)ap)[0];
    a.q[1] = ((const uint4*)ap)[2];
#pragma unroll
    for (int j = 0; j < 4; ++j) {
      Frag16 b;
      loadBf(b, wf, (size_t)kt * NT + tn0 + j, lane);
      acc[j] = __builtin_amdgcn_wmma_f32_16x16x32_bf16(false, a.v, false, b.v,
                                                       (short)0, acc[j],
                                                       false, false);
    }
  }
  // epilogue: 64-wide strip lies in exactly one of q/k/v and one head
  const int wch = n0 >> 10;
  const int h   = (n0 & (C_ - 1)) >> 6;
  const int bb  = m0 >> 11;
  const int ns0 = m0 & (N_ - 1);
  unsigned short* dst = (wch == 0) ? Qh : (wch == 1) ? Kh : Vh;
  const float sc = (wch == 0) ? SCALE_ : 1.0f;
  unsigned short* obase = dst + ((size_t)(bb * H_ + h) * N_ + ns0) * D_;
#pragma unroll
  for (int j = 0; j < 4; ++j)
#pragma unroll
    for (int r = 0; r < 8; ++r)
      obase[(size_t)(r + (hv << 3)) * D_ + j * 16 + ln] = f2bf(acc[j][r] * sc);
}

// ---------------------------------------------------------------------------
// Attention: one workgroup per (b, h, 16-query-row tile).
// Dynamic LDS: 128KB fp32 scores + 64KB bf16 probs (CDNA5 320KB WGP LDS).
__global__ __launch_bounds__(256) void attn_kernel(
    const unsigned short* __restrict__ Qh,
    const unsigned short* __restrict__ Kh,
    const unsigned short* __restrict__ Vf,
    unsigned short* __restrict__ Ob) {
  extern __shared__ char smem[];
  float*          Sbuf = (float*)smem;                      // [16][N_]
  unsigned short* Pb   = (unsigned short*)(smem + 16 * N_ * sizeof(float));

  const int bid  = blockIdx.x;
  const int qt   = bid & 127;
  const int h    = (bid >> 7) & (H_ - 1);
  const int b    = bid >> 11;
  const int bh   = b * H_ + h;
  const int m0   = qt * 16;
  const int wave = threadIdx.x >> 5;
  const int lane = threadIdx.x & 31;
  const int hv   = lane >> 4;
  const int ln   = lane & 15;

  // ---- Phase 1: S = (Q*scale) * K^T ; A fragments loaded once ----
  Frag16 a0, a1;
  {
    const unsigned short* qrow = Qh + ((size_t)bh * N_ + m0 + ln) * D_;
    loadA(a0, qrow, 0, hv);
    loadA(a1, qrow, 32, hv);
  }
  for (int kt = wave; kt < N_ / 16; kt += 8) {
    Frag16 b0, b1;
    const unsigned short* krow = Kh + ((size_t)bh * N_ + kt * 16 + ln) * D_;
    loadA(b0, krow, 0, hv);     // B[k=d][n=key]: K rows contiguous in d
    loadA(b1, krow, 32, hv);
    v8f s = {};
    s = __builtin_amdgcn_wmma_f32_16x16x32_bf16(false, a0.v, false, b0.v,
                                                (short)0, s, false, false);
    s = __builtin_amdgcn_wmma_f32_16x16x32_bf16(false, a1.v, false, b1.v,
                                                (short)0, s, false, false);
#pragma unroll
    for (int r = 0; r < 8; ++r)
      Sbuf[(r + (hv << 3)) * N_ + kt * 16 + ln] = s[r];
  }
  __syncthreads();

  // ---- Phase 2: full-row softmax (16 threads/row); probs -> LDS bf16 ----
  {
    const int row = threadIdx.x >> 4;
    const int i   = threadIdx.x & 15;
    float* Srow = Sbuf + row * N_;
    unsigned short* Prow = Pb + row * N_;
    float mx = -3.0e38f;
    for (int j = i; j < N_; j += 16) mx = fmaxf(mx, Srow[j]);
#pragma unroll
    for (int o = 8; o >= 1; o >>= 1) mx = fmaxf(mx, __shfl_xor(mx, o, 16));
    float sum = 0.0f;
    for (int j = i; j < N_; j += 16) {
      const float p = __expf(Srow[j] - mx);
      Srow[j] = p;
      sum += p;
    }
#pragma unroll
    for (int o = 8; o >= 1; o >>= 1) sum += __shfl_xor(sum, o, 16);
    const float inv = 1.0f / sum;
    for (int j = i; j < N_; j += 16) Prow[j] = f2bf(Srow[j] * inv);
  }
  __syncthreads();

  // ---- Phase 3: O = P (16x2048) * V (2048x64); waves 0..3 take 16 cols ----
  if (wave < 4) {
    v8f o = {};
    const unsigned short* prow = Pb + (size_t)ln * N_;
    for (int kt = 0; kt < N_ / 32; ++kt) {
      Frag16 a, bb;
      const unsigned short* p = prow + kt * 32 + (hv << 3);
      a.q[0] = ((const uint4*)p)[0];          // 2 x ds_load_b128
      a.q[1] = ((const uint4*)p)[2];
      loadBf(bb, Vf, ((size_t)bh * (N_ / 32) + kt) * 4 + wave, lane);
      o = __builtin_amdgcn_wmma_f32_16x16x32_bf16(false, a.v, false, bb.v,
                                                  (short)0, o, false, false);
    }
    unsigned short* orow =
        Ob + ((size_t)b * N_ + m0) * C_ + h * D_ + wave * 16 + ln;
#pragma unroll
    for (int r = 0; r < 8; ++r)
      orow[(size_t)(r + (hv << 3)) * C_] = f2bf(o[r]);
  }
}

// ---------------------------------------------------------------------------
// Output projection: block = 16 rows x 512 cols; LDS A panel; 16x64 per wave.
__global__ __launch_bounds__(256) void proj_gemm(
    const unsigned short* __restrict__ Ab,
    const unsigned short* __restrict__ Wf,
    const float* __restrict__ bias,
    float* __restrict__ out) {
  __shared__ unsigned short Apan[16 * C_];         // 32 KB
  const int tid = threadIdx.x;
  const int grp = blockIdx.x & 1;                  // 1024 / 512 col groups
  const int tm  = blockIdx.x >> 1;
  const int m0  = tm * 16;

  stage_panel(Ab + (size_t)m0 * C_, Apan, (16 * C_) / 8, tid, 256);
  __syncthreads();

  const int wave = tid >> 5, lane = tid & 31;
  const int hv = lane >> 4, ln = lane & 15;
  const int n0  = grp * 512 + wave * 64;
  const int NT  = C_ / 16;                         // 64
  const int tn0 = n0 >> 4;

  v8f acc[4] = {};
  for (int kt = 0; kt < C_ / 32; ++kt) {
    Frag16 a;
    const unsigned short* ap = Apan + ln * C_ + kt * 32 + (hv << 3);
    a.q[0] = ((const uint4*)ap)[0];
    a.q[1] = ((const uint4*)ap)[2];
#pragma unroll
    for (int j = 0; j < 4; ++j) {
      Frag16 b;
      loadBf(b, Wf, (size_t)kt * NT + tn0 + j, lane);
      acc[j] = __builtin_amdgcn_wmma_f32_16x16x32_bf16(false, a.v, false, b.v,
                                                       (short)0, acc[j],
                                                       false, false);
    }
  }
  float* orow = out + (size_t)m0 * C_ + n0;
#pragma unroll
  for (int j = 0; j < 4; ++j) {
    const float bi = bias[n0 + j * 16 + ln];
#pragma unroll
    for (int r = 0; r < 8; ++r)
      orow[(size_t)(r + (hv << 3)) * C_ + j * 16 + ln] = acc[j][r] + bi;
  }
}

// ---------------------------------------------------------------------------
extern "C" void kernel_launch(void* const* d_in, const int* in_sizes, int n_in,
                              void* d_out, int out_size, void* d_ws,
                              size_t ws_size, hipStream_t stream) {
  const float* x      = (const float*)d_in[0];   // (4,2048,1024)
  const float* w_qkv  = (const float*)d_in[1];   // (1024,3072)
  const float* w_proj = (const float*)d_in[2];   // (1024,1024)
  const float* b_proj = (const float*)d_in[3];   // (1024,)
  float* out = (float*)d_out;

  char* ws = (char*)d_ws;
  const size_t MB = 1u << 20;
  unsigned short* xb  = (unsigned short*)(ws + 0 * MB);    // 16 MiB
  unsigned short* wqf = (unsigned short*)(ws + 16 * MB);   //  6 MiB (frag)
  unsigned short* wpf = (unsigned short*)(ws + 22 * MB);   //  2 MiB (frag)
  unsigned short* Qh  = (unsigned short*)(ws + 24 * MB);   // 16 MiB
  unsigned short* Kh  = (unsigned short*)(ws + 40 * MB);   // 16 MiB
  unsigned short* Vh  = (unsigned short*)(ws + 56 * MB);   // 16 MiB
  unsigned short* Vf  = (unsigned short*)(ws + 72 * MB);   // 16 MiB (frag)
  unsigned short* Ab  = (unsigned short*)(ws + 88 * MB);   // 16 MiB

  // 1) conversions / weight repacks (one-shot, HBM-bound)
  cvt_bf16<<<4096, 256, 0, stream>>>(x, xb, B_ * N_ * C_);
  repack_w_frag<<<4096, 256, 0, stream>>>(w_qkv, wqf, C_, 3 * C_);
  repack_w_frag<<<2048, 256, 0, stream>>>(w_proj, wpf, C_, C_);

  // 2) QKV projection: 512 row tiles x 6 column groups
  qkv_gemm<<<512 * 6, 256, 0, stream>>>(xb, wqf, Qh, Kh, Vh);

  // 3) V -> fragment-major for P*V
  repack_v<<<4096, 256, 0, stream>>>(Vh, Vf);

  // 4) Attention: one block per (b,h,16 query rows); 192 KB dynamic LDS
  attn_kernel<<<B_ * H_ * (N_ / 16), 256,
                16 * N_ * (sizeof(float) + sizeof(unsigned short)), stream>>>(
      Qh, Kh, Vf, Ab);

  // 5) Output projection + bias: 512 row tiles x 2 column groups
  proj_gemm<<<512 * 2, 256, 0, stream>>>(Ab, wpf, b_proj, out);
}
